// SMN_15564961480819
// MI455X (gfx1250) — matrix operand order repaired and checked by reading, not compiled, not run on hardware
//
#include <hip/hip_runtime.h>

typedef _Float16 half_t;
typedef __attribute__((ext_vector_type(8)))  _Float16 v8h;
typedef __attribute__((ext_vector_type(16))) _Float16 v16h;
typedef __attribute__((ext_vector_type(8)))  float    v8f;

#define NSET  64
#define NITEM 128
#define DIM   64
#define DM    128   // HEADS*HS

// ---------------------------------------------------------------------------
// WMMA fragment gathers.  All matrices consumed as GEMM operands are stored
// contiguous along K:  A as [M][K] row-major, B as B^T = [N][K] row-major.
// Layouts per CDNA5 ISA 7.12.2 (16-bit A 16x32, 16-bit B 32x16, f32 C 16x16).
// ---------------------------------------------------------------------------
template <typename P>
__device__ __forceinline__ v16h frag_a(const P* base, int ld, int row0, int col0, int lane) {
    const P* p = base + (size_t)(row0 + (lane & 15)) * ld + col0 + ((lane >> 4) << 3);
    v8h lo = *(const v8h*)(p);        // K = k0 + 8*(lane>>4)      .. +7
    v8h hi = *(const v8h*)(p + 16);   // K = k0 + 16 + 8*(lane>>4) .. +7
    return __builtin_shufflevector(lo, hi, 0,1,2,3,4,5,6,7,8,9,10,11,12,13,14,15);
}
template <typename P>
__device__ __forceinline__ v16h frag_b(const P* base, int ld, int ncol0, int k0, int lane) {
    const P* p = base + (size_t)(ncol0 + (lane & 15)) * ld + k0 + ((lane >> 4) << 3);
    v8h lo = *(const v8h*)(p);        // K = k0 + 8*(lane>>4)      .. +7
    v8h hi = *(const v8h*)(p + 16);   // K = k0 + 16 + 8*(lane>>4) .. +7
    return __builtin_shufflevector(lo, hi, 0,1,2,3,4,5,6,7,8,9,10,11,12,13,14,15);
}

// ---------------------------------------------------------------------------
// Weight packing: f32 -> f16 B^T layouts.
//   WqkvT[384][64] : n<128 Wq, n<256 Wk, n<384 Wv  (each W is [64][128])
//   WhT  [64][128] : Wh is [128][64]
//   WcT  [128][64] : Wc is [64][128]
// ---------------------------------------------------------------------------
__global__ void pack_weights(const float* __restrict__ Wq, const float* __restrict__ Wk,
                             const float* __restrict__ Wv, const float* __restrict__ Wh,
                             const float* __restrict__ Wc,
                             half_t* __restrict__ WqkvT, half_t* __restrict__ WhT,
                             half_t* __restrict__ WcT) {
    int i = blockIdx.x * blockDim.x + threadIdx.x;
    if (i < 384 * 64) {
        int n = i >> 6, k = i & 63;
        const float* W = (n < 128) ? Wq : ((n < 256) ? Wk : Wv);
        WqkvT[i] = (half_t)W[k * 128 + (n & 127)];
    } else if (i < 384 * 64 + 64 * 128) {
        int j = i - 24576;
        int n = j >> 7, k = j & 127;
        WhT[j] = (half_t)Wh[k * 64 + n];
    } else if (i < 384 * 64 + 64 * 128 + 128 * 64) {
        int j = i - 32768;
        int n = j >> 6, k = j & 63;
        WcT[j] = (half_t)Wc[k * 128 + n];
    }
}

// ---------------------------------------------------------------------------
// Masked set-normalization (exact reference semantics), f32 -> f16.
// One block per set; first 64 blocks = x, next 64 = y.
// ---------------------------------------------------------------------------
__global__ void setnorm_kernel(const float* __restrict__ x, const float* __restrict__ y,
                               const int* __restrict__ xsz, const int* __restrict__ ysz,
                               half_t* __restrict__ xn, half_t* __restrict__ yn) {
    int blk = blockIdx.x;
    int set = (blk < NSET) ? blk : blk - NSET;
    const float* src = (blk < NSET) ? (x + (size_t)set * 8192) : (y + (size_t)set * 8192);
    const int*   szp = (blk < NSET) ? xsz : ysz;
    half_t*      dst = (blk < NSET) ? (xn + (size_t)set * 8192) : (yn + (size_t)set * 8192);

    __shared__ float red[256];
    __shared__ float rowmask[128];
    __shared__ float stats[2];
    int tid = threadIdx.x;

    if (tid < 128) {
        float rs = 0.f;
        for (int f = 0; f < 64; ++f) rs += src[tid * 64 + f];
        rowmask[tid] = (rs != 0.f) ? 1.f : 0.f;
    }
    float ts = 0.f;
    for (int i = tid; i < 8192; i += 256) ts += src[i];
    red[tid] = ts;
    __syncthreads();
    for (int s = 128; s > 0; s >>= 1) { if (tid < s) red[tid] += red[tid + s]; __syncthreads(); }
    float denom = (float)szp[set] * 64.f;
    if (tid == 0) stats[0] = red[0] / denom;
    __syncthreads();
    float mean = stats[0];

    float vs = 0.f;
    for (int i = tid; i < 8192; i += 256) {
        float d = src[i] - mean;
        vs += d * d * rowmask[i >> 6];
    }
    red[tid] = vs;
    __syncthreads();
    for (int s = 128; s > 0; s >>= 1) { if (tid < s) red[tid] += red[tid + s]; __syncthreads(); }
    if (tid == 0) stats[1] = 1.f / (sqrtf(red[0] / denom) + 1e-8f);
    __syncthreads();
    float inv = stats[1];

    for (int i = tid; i < 8192; i += 256) {
        float o = (src[i] - mean) * inv * rowmask[i >> 6];
        dst[i] = (half_t)o;
    }
}

// ---------------------------------------------------------------------------
// Generic batched WMMA GEMM:  C[z] = A[z] (MxK) @ B[z] (KxN, stored as B^T[N][K]).
// One wave per 16x16 output tile; K stepped by 32 via v_wmma_f32_16x16x32_f16.
// Batch offsets decomposed as z = zo*innerN + zi with per-operand strides.
// mode 0: plain f16 store.  mode 1: QKV epilogue (n<256 -> qk[z][128][256],
//                                   n>=256 -> vT[z][128][128] transposed).
// ---------------------------------------------------------------------------
__global__ void gemm_wmma(const half_t* __restrict__ A, int lda, int aO, int aI,
                          const half_t* __restrict__ Bt, int ldb, int bO, int bI,
                          half_t* __restrict__ C, int ldc, int cO, int cI,
                          half_t* __restrict__ C2,
                          int tilesN, int K, int innerN, int mode) {
    int lane = threadIdx.x & 31, wave = threadIdx.x >> 5;
    int g = blockIdx.x * 4 + wave;
    int tm = g / tilesN, tn = g % tilesN;
    int z = blockIdx.z;
    int zo = z / innerN, zi = z % innerN;

    const half_t* Ab = A + (size_t)zo * aO + (size_t)zi * aI;
    const half_t* Bb = Bt + (size_t)zo * bO + (size_t)zi * bI;

    v8f c = {};
    for (int k0 = 0; k0 < K; k0 += 32) {
        v16h av = frag_a(Ab, lda, tm * 16, k0, lane);
        v16h bv = frag_b(Bb, ldb, tn * 16, k0, lane);
        c = __builtin_amdgcn_wmma_f32_16x16x32_f16(false, av, false, bv, (short)0, c, false, false);
    }

    int n  = tn * 16 + (lane & 15);
    int mb = tm * 16 + ((lane >> 4) << 3);
    if (mode == 0) {
        half_t* Cp = C + (size_t)zo * cO + (size_t)zi * cI;
#pragma unroll
        for (int r = 0; r < 8; ++r) Cp[(size_t)(mb + r) * ldc + n] = (half_t)c[r];
    } else {
        if (n < 256) {
            half_t* Cp = C + (size_t)z * 32768;
#pragma unroll
            for (int r = 0; r < 8; ++r) Cp[(size_t)(mb + r) * 256 + n] = (half_t)c[r];
        } else {
            half_t* Cp = C2 + (size_t)z * 16384 + (size_t)(n - 256) * 128;
#pragma unroll
            for (int r = 0; r < 8; ++r) Cp[mb + r] = (half_t)c[r];
        }
    }
}

// ---------------------------------------------------------------------------
// Fused scores + masked softmax for one (set, head, 16-row band).
// 8 waves -> 16x128 score band in LDS (f32); mask = (s != 0) as in reference.
// qk layout: [set][item][256] (q cols 0..127, k cols 128..255).
// attn out:  [set][head][128][128] f16.
// ---------------------------------------------------------------------------
__global__ void scores_softmax(const half_t* __restrict__ qk, half_t* __restrict__ attn) {
    int band = blockIdx.x, h = blockIdx.y, set = blockIdx.z;
    const half_t* base = qk + (size_t)set * 32768;
    const half_t* Aq = base;                // lda 256, col offset h*64
    const half_t* Bk = base + 128 + h * 64; // B^T rows = items, ldb 256

    int lane = threadIdx.x & 31, wave = threadIdx.x >> 5;
    __shared__ float sc[16][128];
    __shared__ float inv[16];

    v8f c = {};
#pragma unroll
    for (int k0 = 0; k0 < 64; k0 += 32) {
        v16h av = frag_a(Aq, 256, band * 16, h * 64 + k0, lane);
        v16h bv = frag_b(Bk, 256, wave * 16, k0, lane);
        c = __builtin_amdgcn_wmma_f32_16x16x32_f16(false, av, false, bv, (short)0, c, false, false);
    }
#pragma unroll
    for (int r = 0; r < 8; ++r)
        sc[r + ((lane >> 4) << 3)][wave * 16 + (lane & 15)] = c[r] * 0.125f;  // /sqrt(64)
    __syncthreads();

    if (threadIdx.x < 16) {
        int r = threadIdx.x;
        float mx = -3.4e38f;
        for (int j = 0; j < 128; ++j) mx = fmaxf(mx, sc[r][j]);
        float s = 0.f;
        for (int j = 0; j < 128; ++j) {
            float v = sc[r][j];
            float e = (v != 0.f) ? __expf(v - mx) : 0.f;
            sc[r][j] = e;
            s += e;
        }
        inv[r] = 1.f / (s + 1e-10f);
    }
    __syncthreads();

    half_t* out = attn + ((size_t)set * 2 + h) * 16384 + (size_t)band * 2048;
    for (int i = threadIdx.x; i < 2048; i += 256) {
        int r = i >> 7, j = i & 127;
        out[i] = (half_t)(sc[r][j] * inv[r]);
    }
}

// ---------------------------------------------------------------------------
// Cross-set bilinear score: one block per (a,b) pair.
//  - Stage xp[a], yp[b] ([128][128] f16) into padded LDS with CDNA5 async
//    copies (global_load_async_to_lds_b128 + s_wait_asynccnt).
//  - Each wave owns (head, row-band) tasks: A fragments stay in VGPRs while
//    sweeping the 8 column tiles (2 WMMAs each), halving LDS gather traffic.
//  - LeakyReLU(0.3) + 1/sqrt(HS) + full reduction via LDS float atomics.
//   out[a][b] = sum_h ( sum leaky(xp_a @ yp_b^T) / 8 ) / (xsz[b]*128) * Wc2[h]
// ---------------------------------------------------------------------------
#define LDS_LD 136   // 272B row stride: 16B aligned, conflict-free b128 gathers
__global__ void cross_score_kernel(const half_t* __restrict__ xp, const half_t* __restrict__ yp,
                                   const int* __restrict__ x_size, const float* __restrict__ Wc2,
                                   float* __restrict__ out) {
    extern __shared__ char smem[];
    half_t* xa = (half_t*)smem;
    half_t* ya = xa + 128 * LDS_LD;

    int b = blockIdx.x, a = blockIdx.y;
    int tid = threadIdx.x;
    const half_t* gx = xp + (size_t)a * 16384;
    const half_t* gy = yp + (size_t)b * 16384;

    // Async staging: 2 arrays * 2048 16-byte chunks, 16 per thread.
    // LDS byte address = low 32 bits of the generic pointer (aperture mapping).
    for (int i = tid; i < 4096; i += 256) {
        int arr = i >> 11;
        int c8  = i & 2047;
        int row = c8 >> 4, col = (c8 & 15) << 3;
        const half_t* src = (arr ? gy : gx) + row * 128 + col;
        half_t*      dstp = (arr ? ya : xa) + row * LDS_LD + col;
        unsigned lds_off = (unsigned)(size_t)dstp;
        asm volatile("global_load_async_to_lds_b128 %0, %1, off"
                     :: "v"(lds_off), "v"(src)
                     : "memory");
    }
    asm volatile("s_wait_asynccnt 0x0" ::: "memory");
    __syncthreads();

    int lane = tid & 31, wave = tid >> 5;
    float acc0 = 0.f, acc1 = 0.f;
    for (int t = wave; t < 16; t += 8) {       // 2 heads * 8 row-bands
        int head = t >> 3, ti = t & 7;
        // A fragments for this (head, row-band): live in VGPRs for the sweep
        v16h a0 = frag_a(xa, LDS_LD, ti * 16, head * 64, lane);
        v16h a1 = frag_a(xa, LDS_LD, ti * 16, head * 64 + 32, lane);
        float s = 0.f;
#pragma unroll
        for (int tj = 0; tj < 8; ++tj) {
            v16h b0 = frag_b(ya, LDS_LD, tj * 16, head * 64, lane);
            v16h b1 = frag_b(ya, LDS_LD, tj * 16, head * 64 + 32, lane);
            v8f c = {};
            c = __builtin_amdgcn_wmma_f32_16x16x32_f16(false, a0, false, b0, (short)0, c, false, false);
            c = __builtin_amdgcn_wmma_f32_16x16x32_f16(false, a1, false, b1, (short)0, c, false, false);
#pragma unroll
            for (int r = 0; r < 8; ++r) {
                float v = c[r];
                s += (v >= 0.f) ? v : 0.3f * v;    // LeakyReLU(0.3)
            }
        }
        if (head == 0) acc0 += s; else acc1 += s;
    }

    __shared__ float red[2];
    if (tid < 2) red[tid] = 0.f;
    __syncthreads();
    atomicAdd(&red[0], acc0 * 0.125f);             // /sqrt(HS)
    atomicAdd(&red[1], acc1 * 0.125f);
    __syncthreads();
    if (tid == 0) {
        float inv = 1.f / ((float)x_size[b] * 128.f);   // /nItem[b]/Iy
        out[a * 64 + b] = red[0] * inv * Wc2[0] + red[1] * inv * Wc2[1];
    }
}

// ---------------------------------------------------------------------------
extern "C" void kernel_launch(void* const* d_in, const int* in_sizes, int n_in,
                              void* d_out, int out_size, void* d_ws, size_t ws_size,
                              hipStream_t stream) {
    const float* x   = (const float*)d_in[0];
    const float* y   = (const float*)d_in[1];
    const float* Wq  = (const float*)d_in[2];
    const float* Wk  = (const float*)d_in[3];
    const float* Wv  = (const float*)d_in[4];
    const float* Wh  = (const float*)d_in[5];
    const float* Wc  = (const float*)d_in[6];
    const float* Wc2 = (const float*)d_in[7];
    const int*   xsz = (const int*)d_in[8];
    const int*   ysz = (const int*)d_in[9];
    float* out = (float*)d_out;

    char* ws = (char*)d_ws;
    size_t off = 0;
    auto carve = [&](size_t bytes) { char* p = ws + off; off += (bytes + 255) & ~(size_t)255; return p; };

    half_t* WqkvT = (half_t*)carve(384 * 64 * 2);
    half_t* WhT   = (half_t*)carve(64 * 128 * 2);
    half_t* WcT   = (half_t*)carve(128 * 64 * 2);
    half_t* xn    = (half_t*)carve((size_t)NSET * 8192 * 2);
    half_t* yn    = (half_t*)carve((size_t)NSET * 8192 * 2);
    half_t* qk_x  = (half_t*)carve((size_t)NSET * 128 * 256 * 2);
    half_t* qk_y  = (half_t*)carve((size_t)NSET * 128 * 256 * 2);
    half_t* vT_x  = (half_t*)carve((size_t)NSET * 128 * 128 * 2);
    half_t* vT_y  = (half_t*)carve((size_t)NSET * 128 * 128 * 2);
    half_t* at_x  = (half_t*)carve((size_t)NSET * 2 * 128 * 128 * 2);
    half_t* at_y  = (half_t*)carve((size_t)NSET * 2 * 128 * 128 * 2);
    half_t* oh_x  = (half_t*)carve((size_t)NSET * 128 * 128 * 2);
    half_t* oh_y  = (half_t*)carve((size_t)NSET * 128 * 128 * 2);
    half_t* hx    = (half_t*)carve((size_t)NSET * 128 * 64 * 2);
    half_t* hy    = (half_t*)carve((size_t)NSET * 128 * 64 * 2);
    half_t* xp    = (half_t*)carve((size_t)NSET * 128 * 128 * 2);
    half_t* yp    = (half_t*)carve((size_t)NSET * 128 * 128 * 2);
    (void)ws_size; (void)in_sizes; (void)n_in; (void)out_size;

    pack_weights<<<160, 256, 0, stream>>>(Wq, Wk, Wv, Wh, Wc, WqkvT, WhT, WcT);
    setnorm_kernel<<<128, 256, 0, stream>>>(x, y, xsz, ysz, xn, yn);

    // QKV projection: [128x64] @ [64x384]  (N tiles=24, K=64), mode 1 epilogue
    gemm_wmma<<<dim3(48, 1, NSET), 128, 0, stream>>>(xn, 64, 8192, 0, WqkvT, 64, 0, 0,
                                                     qk_x, 0, 0, 0, vT_x, 24, 64, 1, 1);
    gemm_wmma<<<dim3(48, 1, NSET), 128, 0, stream>>>(yn, 64, 8192, 0, WqkvT, 64, 0, 0,
                                                     qk_y, 0, 0, 0, vT_y, 24, 64, 1, 1);

    // scores + masked softmax per (band, head, set)
    scores_softmax<<<dim3(8, 2, NSET), 256, 0, stream>>>(qk_x, at_x);
    scores_softmax<<<dim3(8, 2, NSET), 256, 0, stream>>>(qk_y, at_y);

    // attn @ v : z over (set,head)=128; C into concat-head layout [set][item][h*64+f]
    gemm_wmma<<<dim3(8, 1, 2 * NSET), 128, 0, stream>>>(at_x, 128, 32768, 16384,
                                                        vT_x, 128, 16384, 8192,
                                                        oh_x, 128, 16384, 64,
                                                        nullptr, 4, 128, 2, 0);
    gemm_wmma<<<dim3(8, 1, 2 * NSET), 128, 0, stream>>>(at_y, 128, 32768, 16384,
                                                        vT_y, 128, 16384, 8192,
                                                        oh_y, 128, 16384, 64,
                                                        nullptr, 4, 128, 2, 0);

    // @ Wh : [128x128] @ [128x64]
    gemm_wmma<<<dim3(8, 1, NSET), 128, 0, stream>>>(oh_x, 128, 16384, 0, WhT, 128, 0, 0,
                                                    hx, 64, 8192, 0, nullptr, 4, 128, 1, 0);
    gemm_wmma<<<dim3(8, 1, NSET), 128, 0, stream>>>(oh_y, 128, 16384, 0, WhT, 128, 0, 0,
                                                    hy, 64, 8192, 0, nullptr, 4, 128, 1, 0);

    // @ Wc : [128x64] @ [64x128]
    gemm_wmma<<<dim3(16, 1, NSET), 128, 0, stream>>>(hx, 64, 8192, 0, WcT, 64, 0, 0,
                                                     xp, 128, 16384, 0, nullptr, 8, 64, 1, 0);
    gemm_wmma<<<dim3(16, 1, NSET), 128, 0, stream>>>(hy, 64, 8192, 0, WcT, 64, 0, 0,
                                                     yp, 128, 16384, 0, nullptr, 8, 64, 1, 0);

    // cross-set bilinear scores (dominant: 4096 pairs x 256 WMMAs)
    int lds_bytes = 2 * 128 * LDS_LD * 2;  // 69632 B, fine on 320KB-LDS WGP
    hipFuncSetAttribute((const void*)cross_score_kernel,
                        hipFuncAttributeMaxDynamicSharedMemorySize, lds_bytes);
    cross_score_kernel<<<dim3(64, 64), 256, lds_bytes, stream>>>(xp, yp, xsz, Wc2, out);
}
